// MetaFunUpdaterLocal_48885317763382
// MI455X (gfx1250) — matrix operand order repaired and checked by reading, not compiled, not run on hardware
//
#include <hip/hip_runtime.h>
#include <hip/hip_bf16.h>

typedef _Float16 half_t;
typedef __attribute__((ext_vector_type(16))) _Float16 v16h;
typedef __attribute__((ext_vector_type(8)))  _Float16 v8h;
typedef __attribute__((ext_vector_type(4)))  _Float16 v4h;
typedef __attribute__((ext_vector_type(8)))  float    v8f;
typedef __attribute__((ext_vector_type(4)))  float    v4f;

// ---- LDS layout (in halves) ----
#define SF_STRIDE 164                       // features rows [128][160] padded
#define SH_STRIDE 132                       // hidden rows [128][128] padded
#define SU_STRIDE 264                       // U^T rows [64][256] padded
#define SH1_OFF   (128 * SF_STRIDE)
#define SU_OFF    (SH1_OFF + 128 * SH_STRIDE)
#define SMEM_HALVES (SU_OFF + 64 * SU_STRIDE)
#define SMEM_BYTES  (SMEM_HALVES * 2)       // 109568 bytes

static __device__ inline v8f wmma_f16(v16h a, v16h b, v8f c) {
  // D = A(16x32 f16) * B(32x16 f16) + C(f32)
  return __builtin_amdgcn_wmma_f32_16x16x32_f16(false, a, false, b, (short)0, c,
                                                false, false);
}

// A fragment from row-major f16 buffer.
// p = base + (m0+ln)*stride + k0 + 8*hi   (ln=lane&15, hi=lane>>4)
// halves 0..7 -> K = 8*hi + 0..7 ; halves 8..15 -> K = 16 + 8*hi + 0..7
static __device__ inline v16h load_a_rm(const half_t* p) {
  union { v16h v; v4h q[4]; } a;
  a.q[0] = *(const v4h*)(p + 0);
  a.q[1] = *(const v4h*)(p + 4);
  a.q[2] = *(const v4h*)(p + 16);
  a.q[3] = *(const v4h*)(p + 20);
  return a.v;
}

// A fragment from row-major f32 buffer with on-the-fly convert.
static __device__ inline v16h load_a_f32(const float* p) {
  union { v16h v; v4h q[4]; } a;
  a.q[0] = __builtin_convertvector(*(const v4f*)(p + 0),  v4h);
  a.q[1] = __builtin_convertvector(*(const v4f*)(p + 4),  v4h);
  a.q[2] = __builtin_convertvector(*(const v4f*)(p + 16), v4h);
  a.q[3] = __builtin_convertvector(*(const v4f*)(p + 20), v4h);
  return a.v;
}

// B fragment from K-contiguous ("[n][k]" transposed) f16 buffer.
// p = base + (n0+ln)*stride + k0 + 8*hi ; halves 0..7 -> K = 8*hi + 0..7 ; halves 8..15 -> K = 16 + 8*hi + 0..7
static __device__ inline v16h load_b_t(const half_t* p) {
  union { v16h v; v8h o[2]; } b;
  b.o[0] = *(const v8h*)(p + 0);
  b.o[1] = *(const v8h*)(p + 16);
  return b.v;
}

// -------- prep: transpose + convert weights to f16 [n][k] in workspace ------
__global__ void prep_weights(const float* __restrict__ W1,
                             const float* __restrict__ W2,
                             const float* __restrict__ W3,
                             half_t* __restrict__ W1T,
                             half_t* __restrict__ W2T,
                             half_t* __restrict__ W3T) {
  int t = blockIdx.x * 256 + threadIdx.x;
  if (t < 160 * 128) { int kk = t >> 7, n = t & 127; W1T[n * 160 + kk] = (half_t)W1[t]; }
  if (t < 128 * 128) { int kk = t >> 7, n = t & 127; W2T[n * 128 + kk] = (half_t)W2[t]; }
  if (t < 128 * 64)  { int kk = t >> 6, n = t & 63;  W3T[n * 128 + kk] = (half_t)W3[t]; }
}

// ------------------------------ fused kernel --------------------------------
// grid = B*C blocks, 256 threads (8 waves). Block (b,k):
//   MLP over j in two 128-row passes -> U^T[e][j] (f16, LDS)
//   einsum: waves 0-3: c_att, waves 4-7: q_att ; out = r - 0.1*delta
__global__ __launch_bounds__(256)
void metafun_fused(const float* __restrict__ x,   const float* __restrict__ y,
                   const float* __restrict__ r_c, const float* __restrict__ r_q,
                   const float* __restrict__ c_att, const float* __restrict__ q_att,
                   const float* __restrict__ b1, const float* __restrict__ b2,
                   const float* __restrict__ b3,
                   const half_t* __restrict__ W1T, const half_t* __restrict__ W2T,
                   const half_t* __restrict__ W3T,
                   float* __restrict__ out_rc, float* __restrict__ out_rq) {
  extern __shared__ half_t smem[];
  half_t* sF  = smem;                 // [128][SF_STRIDE]; reused as sH2 after GEMM1
  half_t* sH1 = smem + SH1_OFF;       // [128][SH_STRIDE]
  half_t* sU  = smem + SU_OFF;        // [64][SU_STRIDE] : U transposed [e][j]

  const int tid  = threadIdx.x;
  const int lane = tid & 31;
  const int ln   = lane & 15;
  const int hi   = lane >> 4;
  const int w    = tid >> 5;          // wave id 0..7
  const int blk  = blockIdx.x;
  const int b    = blk >> 8;
  const int k    = blk & 255;

  const float* xb  = x + (size_t)b * 256 * 64;
  const float* yb  = y + (size_t)b * 256 * 32;
  const float* rcb = r_c + ((size_t)(b * 256 + k)) * 256 * 64;   // [j][e]

  for (int pass = 0; pass < 2; ++pass) {
    const int jbase = pass * 128;

    // ---- build features [128][160] = [x(64) | y(32) | r_c(64)] as f16 ----
    for (int idx = tid; idx < 128 * 40; idx += 256) {
      int row = idx / 40, c = idx - row * 40;
      int j = jbase + row;
      v4f f; int dsto;
      if (c < 16)      { f = *(const v4f*)(xb  + (size_t)j * 64 + 4 * c);        dsto = 4 * c; }
      else if (c < 24) { f = *(const v4f*)(yb  + (size_t)j * 32 + 4 * (c - 16)); dsto = 64 + 4 * (c - 16); }
      else             { f = *(const v4f*)(rcb + (size_t)j * 64 + 4 * (c - 24)); dsto = 96 + 4 * (c - 24); }
      *(v4h*)(sF + row * SF_STRIDE + dsto) = __builtin_convertvector(f, v4h);
    }
    __syncthreads();

    // ---- GEMM1: [128x160] @ W1 -> relu -> sH1 [128x128] ----
    {
      const int n0 = 16 * w;
      const float bias = b1[n0 + ln];
      v16h bfr[5];
#pragma unroll
      for (int kc = 0; kc < 5; ++kc)
        bfr[kc] = load_b_t(W1T + (size_t)(n0 + ln) * 160 + 32 * kc + 8 * hi);
      for (int mt = 0; mt < 8; ++mt) {
        v8f acc = {};
#pragma unroll
        for (int kc = 0; kc < 5; ++kc) {
          v16h a = load_a_rm(sF + (16 * mt + ln) * SF_STRIDE + 32 * kc + 8 * hi);
          acc = wmma_f16(a, bfr[kc], acc);
        }
#pragma unroll
        for (int r = 0; r < 8; ++r) {
          float v = acc[r] + bias;
          v = v > 0.f ? v : 0.f;
          sH1[(16 * mt + r + 8 * hi) * SH_STRIDE + n0 + ln] = (half_t)v;
        }
      }
    }
    __syncthreads();

    // ---- GEMM2: sH1 @ W2 -> relu -> sH2 (aliases sF) ----
    {
      const int n0 = 16 * w;
      const float bias = b2[n0 + ln];
      v16h bfr[4];
#pragma unroll
      for (int kc = 0; kc < 4; ++kc)
        bfr[kc] = load_b_t(W2T + (size_t)(n0 + ln) * 128 + 32 * kc + 8 * hi);
      for (int mt = 0; mt < 8; ++mt) {
        v8f acc = {};
#pragma unroll
        for (int kc = 0; kc < 4; ++kc) {
          v16h a = load_a_rm(sH1 + (16 * mt + ln) * SH_STRIDE + 32 * kc + 8 * hi);
          acc = wmma_f16(a, bfr[kc], acc);
        }
#pragma unroll
        for (int r = 0; r < 8; ++r) {
          float v = acc[r] + bias;
          v = v > 0.f ? v : 0.f;
          sF[(16 * mt + r + 8 * hi) * SH_STRIDE + n0 + ln] = (half_t)v;
        }
      }
    }
    __syncthreads();

    // ---- GEMM3: sH2 @ W3 + b3 -> U^T[e][jbase + ...] ----
    {
      const int e0 = 16 * (w & 3);
      const int mtbase = (w >> 2) * 4;
      const float bias = b3[e0 + ln];
      v16h bfr[4];
#pragma unroll
      for (int kc = 0; kc < 4; ++kc)
        bfr[kc] = load_b_t(W3T + (size_t)(e0 + ln) * 128 + 32 * kc + 8 * hi);
      for (int mi = 0; mi < 4; ++mi) {
        const int mt = mtbase + mi;
        v8f acc = {};
#pragma unroll
        for (int kc = 0; kc < 4; ++kc) {
          v16h a = load_a_rm(sF + (16 * mt + ln) * SH_STRIDE + 32 * kc + 8 * hi);
          acc = wmma_f16(a, bfr[kc], acc);
        }
        v8h o;
#pragma unroll
        for (int r = 0; r < 8; ++r) o[r] = (half_t)(acc[r] + bias);
        // D row m = r + 8*hi are consecutive j -> one 16B store per lane
        *(v8h*)(sU + (e0 + ln) * SU_STRIDE + jbase + 16 * mt + 8 * hi) = o;
      }
    }
    __syncthreads();
  }

  // ---- einsum: delta[i][e] = att[i][j] * U[j][e] ; out = r - 0.1*delta ----
  {
    const int e0  = 16 * (w & 3);
    const int map = w >> 2;                        // 0: c_att/r_c, 1: q_att/r_q
    const float* att  = (map ? q_att : c_att) + (size_t)b * 256 * 256;
    const float* rin  = (map ? r_q : r_c) + ((size_t)(b * 256 + k)) * 256 * 64;
    float*       outp = (map ? out_rq : out_rc) + ((size_t)(b * 256 + k)) * 256 * 64;

    v16h bfr[8];                                   // U^T tiles, reused for all i
#pragma unroll
    for (int kc = 0; kc < 8; ++kc)
      bfr[kc] = load_b_t(sU + (e0 + ln) * SU_STRIDE + 32 * kc + 8 * hi);

    for (int it = 0; it < 16; ++it) {
      const int i0 = 16 * it;
      v8f acc = {};
#pragma unroll
      for (int kc = 0; kc < 8; ++kc) {
        v16h a = load_a_f32(att + (size_t)(i0 + ln) * 256 + 32 * kc + 8 * hi);
        acc = wmma_f16(a, bfr[kc], acc);
      }
#pragma unroll
      for (int r = 0; r < 8; ++r) {
        const int i = i0 + r + 8 * hi;
        const size_t off = (size_t)i * 64 + e0 + ln;
        outp[off] = rin[off] - 0.1f * acc[r];
      }
    }
  }
}

// ------------------------------- launch -------------------------------------
extern "C" void kernel_launch(void* const* d_in, const int* in_sizes, int n_in,
                              void* d_out, int out_size, void* d_ws, size_t ws_size,
                              hipStream_t stream) {
  const float* x     = (const float*)d_in[0];
  const float* y     = (const float*)d_in[1];
  const float* r_c   = (const float*)d_in[2];
  const float* r_q   = (const float*)d_in[3];
  const float* c_att = (const float*)d_in[4];
  const float* q_att = (const float*)d_in[5];
  const float* W1    = (const float*)d_in[6];
  const float* b1    = (const float*)d_in[7];
  const float* W2    = (const float*)d_in[8];
  const float* b2    = (const float*)d_in[9];
  const float* W3    = (const float*)d_in[10];
  const float* b3    = (const float*)d_in[11];

  half_t* W1T = (half_t*)d_ws;                         // 160*128 halves
  half_t* W2T = (half_t*)((char*)d_ws + 40960);        // 128*128 halves
  half_t* W3T = (half_t*)((char*)d_ws + 40960 + 32768);// 64*128 halves

  float* out_rc = (float*)d_out;
  float* out_rq = out_rc + (size_t)4 * 256 * 256 * 64;

  (void)in_sizes; (void)n_in; (void)out_size; (void)ws_size;

  hipFuncSetAttribute((const void*)metafun_fused,
                      hipFuncAttributeMaxDynamicSharedMemorySize, SMEM_BYTES);

  prep_weights<<<80, 256, 0, stream>>>(W1, W2, W3, W1T, W2T, W3T);
  metafun_fused<<<4 * 256, 256, SMEM_BYTES, stream>>>(
      x, y, r_c, r_q, c_att, q_att, b1, b2, b3, W1T, W2T, W3T, out_rc, out_rq);
}
